// ScaledDotProductAttentionLayer_66855460930242
// MI455X (gfx1250) — compile-verified, matrix operand order checked by reading, not run on hardware
//
#include <hip/hip_runtime.h>

#define G_GROUPS 64
#define L_ROWS   1024
#define D_DIM    512
#define QT       16                 // queries per workgroup
#define KT       64                 // keys per tile
#define NT       (L_ROWS / KT)      // 16 tiles
#define NTHREADS 128                // 4 waves -> 2 WGs co-resident per WGP
#define SCALE_F  0.04419417382415922f   // 1/sqrt(512)

typedef __attribute__((ext_vector_type(16))) __bf16 v16bf;
typedef __attribute__((ext_vector_type(8)))  __bf16 v8bf;
typedef __attribute__((ext_vector_type(4)))  __bf16 v4bf;
typedef __attribute__((ext_vector_type(8)))  float  v8f;

// Load one 16x32 bf16 WMMA operand fragment from an LDS tile.
// Works for both A (rows = M) and B (rows = N of a transposed source):
//   lane<16  : row = lane,     k = kbase + {0..7, 16..23}
//   lane>=16 : row = lane-16,  k = kbase + {8..15, 24..31}
__device__ __forceinline__ v16bf load_frag(const __bf16* base, int rowstride,
                                           int lane, int kbase) {
  const int r  = lane & 15;
  const int ko = (lane >> 4) << 3;
  const __bf16* p = base + r * rowstride + kbase + ko;
  union { v16bf v; v8bf h[2]; } u;
  u.h[0] = *(const v8bf*)(p);        // 16B aligned LDS load
  u.h[1] = *(const v8bf*)(p + 16);   // 16B aligned LDS load
  return u.v;
}

__global__ void zero_f32(float* p, int n) {
  int i = blockIdx.x * 256 + threadIdx.x;
  if (i < n) p[i] = 0.0f;
}

__global__ __launch_bounds__(NTHREADS)
void fa_fwd_meanpool(const float* __restrict__ x, float* __restrict__ out) {
  const int wg   = blockIdx.x;
  const int g    = wg >> 6;            // 64 query-tiles per group
  const int qt   = wg & 63;
  const int tid  = threadIdx.x;
  const int wave = tid >> 5;           // 0..3
  const int lane = tid & 31;

  const float* xg = x + (size_t)g * L_ROWS * D_DIM;
  const float* Qg = xg + (size_t)qt * QT * D_DIM;

  // ---- LDS: 163,648 B  (<= 320KB/2 -> two workgroups co-resident per WGP) ----
  __shared__ alignas(16) __bf16 Qs [QT][D_DIM + 8];   // 16,640 B (staging only)
  __shared__ alignas(16) __bf16 Ks [KT][D_DIM + 8];   // 66,560 B (QK^T B-op)
  __shared__ alignas(16) __bf16 KsT[D_DIM][KT + 8];   // 73,728 B (PV   B-op)
  __shared__ alignas(16) __bf16 Ps [QT][KT + 8];      //  2,304 B (probs, bf16)
  __shared__ alignas(16) float  Ss [QT][KT + 2];      //  4,224 B (raw scores)
  __shared__ alignas(16) float  row_max[QT], row_sum[QT], row_scale[QT];

  // ---- init softmax state + stage Q (f32 -> bf16) ----
  if (tid < QT) {
    row_max[tid]   = -__builtin_inff();
    row_sum[tid]   = 0.0f;
    row_scale[tid] = 0.0f;
  }
  for (int i = tid; i < QT * (D_DIM / 4); i += NTHREADS) {  // 16 iters
    const int r = i >> 7;               // D/4 = 128
    const int c = (i & 127) << 2;
    const float4 v = *(const float4*)(Qg + r * D_DIM + c);
    v4bf q; q[0] = (__bf16)v.x; q[1] = (__bf16)v.y;
            q[2] = (__bf16)v.z; q[3] = (__bf16)v.w;
    *(v4bf*)&Qs[r][c] = q;
  }
  __syncthreads();

  // ---- hoist the loop-invariant Q A-fragments into registers (128 VGPRs) ----
  v16bf qf[16];
#pragma unroll
  for (int kc = 0; kc < 16; ++kc)
    qf[kc] = load_frag(&Qs[0][0], D_DIM + 8, lane, kc * 32);

  // per-wave output accumulator: 16 queries x 128 dims (8 x 16x16 f32 frags)
  v8f Oacc[8];
#pragma unroll
  for (int nc = 0; nc < 8; ++nc)
#pragma unroll
    for (int j = 0; j < 8; ++j) Oacc[nc][j] = 0.0f;

  const int rb0 = (lane >> 4) << 3;     // base accumulator row for this lane

  for (int t = 0; t < NT; ++t) {
    __syncthreads();                    // previous tile fully consumed

    // ---- stage K tile in 4x4 blocks: coalesced loads, wide b64 LDS stores ----
    const float* Kt = xg + (size_t)t * KT * D_DIM;
    for (int i = tid; i < (KT / 4) * (D_DIM / 4); i += NTHREADS) { // 16 iters
      const int rb = (i >> 7) << 2;     // key row base  (0..60)
      const int cb = (i & 127) << 2;    // dim  col base (0..508)
      __bf16 tmp[4][4];
#pragma unroll
      for (int rr = 0; rr < 4; ++rr) {
        const float4 v = *(const float4*)(Kt + (rb + rr) * D_DIM + cb);
        tmp[rr][0] = (__bf16)v.x; tmp[rr][1] = (__bf16)v.y;
        tmp[rr][2] = (__bf16)v.z; tmp[rr][3] = (__bf16)v.w;
      }
#pragma unroll
      for (int rr = 0; rr < 4; ++rr) {  // row-major copy (dims contiguous)
        v4bf kv; kv[0]=tmp[rr][0]; kv[1]=tmp[rr][1];
                 kv[2]=tmp[rr][2]; kv[3]=tmp[rr][3];
        *(v4bf*)&Ks[rb + rr][cb] = kv;
      }
#pragma unroll
      for (int cc = 0; cc < 4; ++cc) {  // transposed copy (keys contiguous)
        v4bf kv; kv[0]=tmp[0][cc]; kv[1]=tmp[1][cc];
                 kv[2]=tmp[2][cc]; kv[3]=tmp[3][cc];
        *(v4bf*)&KsT[cb + cc][rb] = kv;
      }
    }
    // prefetch next K tile into the cache hierarchy (global_prefetch_b8)
    if (t + 1 < NT) {
      const float* Kn = xg + (size_t)(t + 1) * KT * D_DIM;
      for (int i = tid; i < 1024; i += NTHREADS)      // 128KB, 128B granules
        __builtin_prefetch(Kn + i * 32, 0, 1);
    }
    __syncthreads();

    // ---- scores: wave w computes S[16q x 16k] for keys [w*16, w*16+16) ----
    {
      v8f s;
#pragma unroll
      for (int j = 0; j < 8; ++j) s[j] = 0.0f;
      const __bf16* kb = &Ks[wave * 16][0];
#pragma unroll
      for (int kc = 0; kc < 16; ++kc) {   // reduce over D = 16 chunks of 32
        const v16bf b = load_frag(kb, D_DIM + 8, lane, kc * 32);
        s = __builtin_amdgcn_wmma_f32_16x16x32_bf16(false, qf[kc], false, b,
                                                    (short)0, s, false, false);
      }
      const int col = wave * 16 + (lane & 15);
#pragma unroll
      for (int j = 0; j < 8; ++j) Ss[rb0 + j][col] = s[j] * SCALE_F;
    }
    __syncthreads();

    // ---- online softmax: 8 threads per query row, shuffle reductions ----
    {
      const int r  = tid >> 3;          // query row 0..15 (8-lane row groups)
      const int sb = tid & 7;
      float v[8], pm = -__builtin_inff();
#pragma unroll
      for (int j = 0; j < 8; ++j) {
        v[j] = Ss[r][sb + 8 * j];
        pm = fmaxf(pm, v[j]);
      }
#pragma unroll
      for (int off = 4; off; off >>= 1)
        pm = fmaxf(pm, __shfl_xor(pm, off, 8));
      const float m_old = row_max[r];
      const float m_new = fmaxf(m_old, pm);
      float ps = 0.0f;
#pragma unroll
      for (int j = 0; j < 8; ++j) {
        const float e = __expf(v[j] - m_new);   // v_exp_f32
        Ps[r][sb + 8 * j] = (__bf16)e;
        ps += e;
      }
#pragma unroll
      for (int off = 4; off; off >>= 1)
        ps += __shfl_xor(ps, off, 8);
      if (sb == 0) {
        const float sc = __expf(m_old - m_new); // exp(-inf)=0 on first tile
        row_scale[r] = sc;
        row_sum[r]   = row_sum[r] * sc + ps;
        row_max[r]   = m_new;
      }
    }
    __syncthreads();

    // ---- PV: wave w owns output dims [w*128, w*128+128) ----
    {
      float fs[8];
#pragma unroll
      for (int j = 0; j < 8; ++j) fs[j] = row_scale[rb0 + j];
#pragma unroll
      for (int nc = 0; nc < 8; ++nc)
#pragma unroll
        for (int j = 0; j < 8; ++j) Oacc[nc][j] *= fs[j];

#pragma unroll
      for (int kc = 0; kc < 2; ++kc) {  // reduce over 64 keys = 2 chunks of 32
        const v16bf a = load_frag(&Ps[0][0], KT + 8, lane, kc * 32);
#pragma unroll
        for (int nc = 0; nc < 8; ++nc) {
          const v16bf b =
              load_frag(&KsT[wave * 128 + nc * 16][0], KT + 8, lane, kc * 32);
          Oacc[nc] = __builtin_amdgcn_wmma_f32_16x16x32_bf16(
              false, a, false, b, (short)0, Oacc[nc], false, false);
        }
      }
    }
  }

  // ---- epilogue: O /= rowsum, mean over the 16 queries, accumulate out ----
  {
    float inv[8];
#pragma unroll
    for (int j = 0; j < 8; ++j) inv[j] = 1.0f / row_sum[rb0 + j];
    const float invL = 1.0f / (float)L_ROWS;
#pragma unroll
    for (int nc = 0; nc < 8; ++nc) {
      float cs = 0.0f;
#pragma unroll
      for (int j = 0; j < 8; ++j) cs += Oacc[nc][j] * inv[j];
      cs += __shfl_xor(cs, 16, 32);     // rows 0..7 + rows 8..15
      if (lane < 16)
        atomicAdd(&out[g * D_DIM + wave * 128 + nc * 16 + lane], cs * invL);
    }
  }
}

extern "C" void kernel_launch(void* const* d_in, const int* in_sizes, int n_in,
                              void* d_out, int out_size, void* d_ws, size_t ws_size,
                              hipStream_t stream) {
  (void)in_sizes; (void)n_in; (void)d_ws; (void)ws_size; (void)out_size;
  const float* x = (const float*)d_in[0];   // d_in[1] = batch_index (unused:
                                            // groups are contiguous, equal-size)
  float* out = (float*)d_out;
  const int n = G_GROUPS * D_DIM;           // 32768
  zero_f32<<<(n + 255) / 256, 256, 0, stream>>>(out, n);
  fa_fwd_meanpool<<<dim3(G_GROUPS * (L_ROWS / QT)), dim3(NTHREADS), 0, stream>>>(x, out);
}
